// TrigramGPT_73289321939517
// MI455X (gfx1250) — matrix operand order repaired and checked by reading, not compiled, not run on hardware
//
#include <hip/hip_runtime.h>
#include <hip/hip_bf16.h>
#include <math.h>

#define VV 32000
#define DD 1024
#define HH 16
#define LL 4
#define BB 2
#define TT 2048
#define DHD 64
#define FF 4096
#define BT (BB * TT)   // 4096 token rows

typedef __attribute__((ext_vector_type(16))) __bf16 v16bf;
typedef __attribute__((ext_vector_type(8)))  __bf16 v8bf;
typedef __attribute__((ext_vector_type(8)))  float  v8f;

// ---------------------------------------------------------------------------
// gfx1250 async global->LDS copy of 16 bytes (ASYNCcnt-tracked), ISA ch.10/14
// ---------------------------------------------------------------------------
__device__ __forceinline__ void async_cp16(unsigned lds_off, const void* gaddr) {
    asm volatile("global_load_async_to_lds_b128 %0, %1, off"
                 :: "v"(lds_off), "v"((unsigned long long)gaddr)
                 : "memory");
}
__device__ __forceinline__ void wait_async0() {
    asm volatile("s_wait_asynccnt 0" ::: "memory");
}

// ---------------------------------------------------------------------------
// Embedding: x[b,t,:] = tok_emb[idx[b,t],:] + pos_emb[t,:]   (f32)
// ---------------------------------------------------------------------------
__global__ void embed_kernel(const int* __restrict__ idx,
                             const float* __restrict__ tok,
                             const float* __restrict__ pos,
                             float* __restrict__ x) {
    int i = blockIdx.x * 256 + threadIdx.x;          // over BT*DD
    if (i >= BT * DD) return;
    int d  = i & (DD - 1);
    int bt = i >> 10;                                 // DD == 1024
    int t  = bt & (TT - 1);
    x[i] = tok[(size_t)idx[bt] * DD + d] + pos[(size_t)t * DD + d];
}

// ---------------------------------------------------------------------------
// Row LayerNorm over D=1024 -> bf16 output (GEMM A operand)
// ---------------------------------------------------------------------------
__global__ __launch_bounds__(256)
void ln_kernel(const float* __restrict__ x, const float* __restrict__ g,
               const float* __restrict__ b, __bf16* __restrict__ out) {
    __shared__ float s1[256];
    __shared__ float s2[256];
    const int tid = threadIdx.x;
    const size_t row = blockIdx.x;
    const float* xr = x + row * DD;
    float sum = 0.f, sum2 = 0.f;
    #pragma unroll
    for (int i = tid; i < DD; i += 256) {
        float v = xr[i];
        sum += v; sum2 += v * v;
    }
    s1[tid] = sum; s2[tid] = sum2;
    __syncthreads();
    for (int s = 128; s > 0; s >>= 1) {
        if (tid < s) { s1[tid] += s1[tid + s]; s2[tid] += s2[tid + s]; }
        __syncthreads();
    }
    const float mean = s1[0] * (1.0f / DD);
    const float var  = s2[0] * (1.0f / DD) - mean * mean;
    const float rstd = rsqrtf(var + 1e-5f);
    #pragma unroll
    for (int i = tid; i < DD; i += 256)
        out[row * DD + i] = (__bf16)((xr[i] - mean) * rstd * g[i] + b[i]);
}

// ---------------------------------------------------------------------------
// Convert + transpose weights: in (K x N, f32 row-major) -> out (N x K, bf16)
// 32x32 LDS tile, both phases coalesced. grid = (N/32, K/32), 256 threads.
// ---------------------------------------------------------------------------
__global__ __launch_bounds__(256)
void convT_kernel(const float* __restrict__ in, __bf16* __restrict__ out,
                  int K, int N) {
    __shared__ float tile[32][33];
    const int n0 = blockIdx.x * 32, k0 = blockIdx.y * 32;
    const int tx = threadIdx.x & 31, ty = threadIdx.x >> 5;   // ty 0..7
    #pragma unroll
    for (int i = 0; i < 4; ++i) {
        int kk = ty + i * 8;
        tile[tx][kk] = in[(size_t)(k0 + kk) * N + n0 + tx];
    }
    __syncthreads();
    #pragma unroll
    for (int i = 0; i < 4; ++i) {
        int nn = ty + i * 8;
        out[(size_t)(n0 + nn) * K + k0 + tx] = (__bf16)tile[nn][tx];
    }
}

// ---------------------------------------------------------------------------
// bf16 WMMA GEMM:  C[M,N] = A[M,K] @ Bt[N,K]^T (+bias[N]) (+resid) (gelu->bf16)
// A: bf16 row-major.  Bt: bf16 pre-transposed (N x K): tiles are verbatim row
// copies staged via global_load_async_to_lds_b128, double-buffered so tile k+1
// copies overlap tile k WMMAs.
// Block tile 128(M) x 256(N) x 32(K), 256 thr (8 waves), wave tile 64x64:
// 16 WMMAs per 16 ds_load_b128 per K-step (1.0 LDS loads per WMMA).
// ---------------------------------------------------------------------------
#define TSTR 40   // LDS row stride in bf16 elements (80B; keeps 16B alignment)

__device__ __forceinline__ void stage_tiles(const __bf16* __restrict__ A,
                                            const __bf16* __restrict__ Bt,
                                            int tm, int tn, int K, int k0,
                                            __bf16* sAb, __bf16* sBb, int tid) {
    // A tile: 128 rows x 32 halves = 512 x 16B chunks (2 per thread)
    #pragma unroll
    for (int i = 0; i < 2; ++i) {
        int c   = tid + i * 256;
        int r   = c >> 2;
        int col = (c & 3) << 3;           // 0,8,16,24 halves
        unsigned la = (unsigned)(unsigned long long)(sAb + r * TSTR + col);
        async_cp16(la, A + (size_t)(tm + r) * K + k0 + col);
    }
    // B tile: 256 rows x 32 halves = 1024 x 16B chunks (4 per thread)
    #pragma unroll
    for (int i = 0; i < 4; ++i) {
        int c   = tid + i * 256;
        int r   = c >> 2;
        int col = (c & 3) << 3;
        unsigned lb = (unsigned)(unsigned long long)(sBb + r * TSTR + col);
        async_cp16(lb, Bt + (size_t)(tn + r) * K + k0 + col);
    }
}

template <bool BIAS, bool RESID, bool GELU_BF>
__global__ __launch_bounds__(256)
void gemm_bf16_wmma(const __bf16* __restrict__ A, const __bf16* __restrict__ Bt,
                    const float* __restrict__ bias, const float* __restrict__ resid,
                    void* __restrict__ Cout, int M, int N, int K) {
    __shared__ alignas(16) __bf16 sA[2][128 * TSTR];   // 2 x 10240 B
    __shared__ alignas(16) __bf16 sB[2][256 * TSTR];   // 2 x 20480 B

    const int tid  = threadIdx.x;
    const int lane = tid & 31;
    const int wave = tid >> 5;
    const int wm   = wave >> 2;     // 0..1 -> 64-row slab
    const int wn   = wave & 3;      // 0..3 -> 64-col slab
    const int lr   = lane & 15;
    const int lhi  = lane >> 4;
    const int tm   = blockIdx.y * 128;
    const int tn   = blockIdx.x * 256;

    v8f acc[4][4] = {};

    // Prologue: async-stage tile 0 into buffer 0
    stage_tiles(A, Bt, tm, tn, K, 0, sA[0], sB[0], tid);

    int buf = 0;
    for (int k0 = 0; k0 < K; k0 += 32, buf ^= 1) {
        wait_async0();        // this wave's copies for tile k0 are done
        __syncthreads();      // all waves staged tile k0; all done with buf^1
        if (k0 + 32 < K)      // overlap next tile's copies with this compute
            stage_tiles(A, Bt, tm, tn, K, k0 + 32, sA[buf ^ 1], sB[buf ^ 1], tid);

        // L2 prefetch two K-tiles ahead (speculative; OOB silently dropped)
        __builtin_prefetch((const void*)(A + (size_t)(tm + (tid >> 1)) * K + k0 + 64), 0, 1);
        __builtin_prefetch((const void*)(Bt + (size_t)(tn + tid) * K + k0 + 64), 0, 1);

        const __bf16* cA = sA[buf];
        const __bf16* cB = sB[buf];

        // B fragments: lane col = bn+lr, element e -> K = lhi*16 + e
        v16bf bfrag[4];
        #pragma unroll
        for (int ns = 0; ns < 4; ++ns) {
            int cc = wn * 64 + ns * 16 + lr;
            v8bf blo = *(const v8bf*)(&cB[cc * TSTR + lhi * 16]);
            v8bf bhi = *(const v8bf*)(&cB[cc * TSTR + lhi * 16 + 8]);
            bfrag[ns] = __builtin_shufflevector(blo, bhi,
                0,1,2,3,4,5,6,7,8,9,10,11,12,13,14,15);
        }
        // A fragments: lane row = am+lr; element e -> K = lhi*16 + e
        #pragma unroll
        for (int ms = 0; ms < 4; ++ms) {
            int rr = wm * 64 + ms * 16 + lr;
            v8bf alo = *(const v8bf*)(&cA[rr * TSTR + lhi * 16]);
            v8bf ahi = *(const v8bf*)(&cA[rr * TSTR + lhi * 16 + 8]);
            v16bf afrag = __builtin_shufflevector(alo, ahi,
                0,1,2,3,4,5,6,7,8,9,10,11,12,13,14,15);
            #pragma unroll
            for (int ns = 0; ns < 4; ++ns) {
                acc[ms][ns] = __builtin_amdgcn_wmma_f32_16x16x32_bf16(
                    false, afrag, false, bfrag[ns],
                    (short)0, acc[ms][ns], false, false);
            }
        }
    }

    // Branch-free epilogue. C/D layout: element g -> row = 8*lhi + g, col = lr
    float*  Cf = (float*)Cout;
    __bf16* Cb = (__bf16*)Cout;
    #pragma unroll
    for (int ms = 0; ms < 4; ++ms) {
        #pragma unroll
        for (int ns = 0; ns < 4; ++ns) {
            const int row0 = tm + wm * 64 + ms * 16 + lhi * 8;
            const int col  = tn + wn * 64 + ns * 16 + lr;
            const float bb = BIAS ? bias[col] : 0.0f;
            size_t off = (size_t)row0 * N + col;
            #pragma unroll
            for (int g = 0; g < 8; ++g) {
                float v = acc[ms][ns][g] + bb;
                if constexpr (RESID) v += resid[off];
                if constexpr (GELU_BF)
                    Cb[off] = (__bf16)(0.5f * v * (1.0f + erff(v * 0.70710678118654752f)));
                else
                    Cf[off] = v;
                off += (size_t)N;
            }
        }
    }
}

// ---------------------------------------------------------------------------
// Width-3 causal band attention: one thread per (b,t,h); bf16 output
// ---------------------------------------------------------------------------
__global__ void band_attn_kernel(const float* __restrict__ q,
                                 const float* __restrict__ k,
                                 const float* __restrict__ v,
                                 __bf16* __restrict__ y) {
    int gid = blockIdx.x * 256 + threadIdx.x;
    if (gid >= BT * HH) return;
    int h  = gid & (HH - 1);
    int bt = gid >> 4;                          // HH == 16
    int t  = bt & (TT - 1);
    const float* qp = q + (size_t)bt * DD + h * DHD;
    const int nv = (t >= 2) ? 3 : (t + 1);
    float s[3];
    float mx = -3.4e38f;
    for (int o = 0; o < nv; ++o) {
        const float* kp = k + (size_t)(bt - o) * DD + h * DHD;
        float acc = 0.f;
        #pragma unroll
        for (int d = 0; d < DHD; ++d) acc += qp[d] * kp[d];
        s[o] = acc * 0.125f;                    // 1/sqrt(64)
        mx = fmaxf(mx, s[o]);
    }
    float p[3] = {0.f, 0.f, 0.f};
    float sum = 0.f;
    for (int o = 0; o < nv; ++o) { p[o] = __expf(s[o] - mx); sum += p[o]; }
    const float inv = 1.0f / sum;
    __bf16* yp = y + (size_t)bt * DD + h * DHD;
    #pragma unroll
    for (int d = 0; d < DHD; ++d) {
        float acc = 0.f;
        for (int o = 0; o < nv; ++o)
            acc += p[o] * v[(size_t)(bt - o) * DD + h * DHD + d];
        yp[d] = (__bf16)(acc * inv);
    }
}

// ---------------------------------------------------------------------------
// Launch
// ---------------------------------------------------------------------------
extern "C" void kernel_launch(void* const* d_in, const int* in_sizes, int n_in,
                              void* d_out, int out_size, void* d_ws, size_t ws_size,
                              hipStream_t stream) {
    const int*   idx   = (const int*)  d_in[0];
    const float* tok   = (const float*)d_in[1];
    const float* pos   = (const float*)d_in[2];
    const float* Wq    = (const float*)d_in[3];
    const float* bq    = (const float*)d_in[4];
    const float* Wk    = (const float*)d_in[5];
    const float* bk    = (const float*)d_in[6];
    const float* Wv    = (const float*)d_in[7];
    const float* bv    = (const float*)d_in[8];
    const float* Wo    = (const float*)d_in[9];
    const float* bo    = (const float*)d_in[10];
    const float* ln1g  = (const float*)d_in[11];
    const float* ln1b  = (const float*)d_in[12];
    const float* W1    = (const float*)d_in[13];
    const float* b1    = (const float*)d_in[14];
    const float* W2    = (const float*)d_in[15];
    const float* b2    = (const float*)d_in[16];
    const float* ln2g  = (const float*)d_in[17];
    const float* ln2b  = (const float*)d_in[18];
    const float* lnfg  = (const float*)d_in[19];
    const float* lnfb  = (const float*)d_in[20];
    const float* lmW   = (const float*)d_in[21];
    float* out = (float*)d_out;

    const size_t SZ = (size_t)BT * DD;              // 4,194,304 elements
    char* base = (char*)d_ws;
    float*  x   = (float*)(base);                   // f32 residual stream
    float*  qf  = (float*)(base + 4  * SZ);
    float*  kf  = (float*)(base + 8  * SZ);
    float*  vf  = (float*)(base + 12 * SZ);
    __bf16* hb  = (__bf16*)(base + 16 * SZ);        // LN out (GEMM A)
    __bf16* yb  = (__bf16*)(base + 18 * SZ);        // attn out (GEMM A)
    __bf16* m1b = (__bf16*)(base + 20 * SZ);        // gelu(h@W1+b1), BT x FF
    __bf16* wt  = (__bf16*)(base + 28 * SZ);        // bf16 transposed weights

    __bf16* wq_t = wt;
    __bf16* wk_t = wt + (size_t)DD * DD;
    __bf16* wv_t = wt + (size_t)2 * DD * DD;
    __bf16* wo_t = wt + (size_t)3 * DD * DD;
    __bf16* w1_t = wt + (size_t)4 * DD * DD;        // FF x DD
    __bf16* w2_t = w1_t + (size_t)DD * FF;          // DD x FF

    const dim3 blk(256);
    const dim3 gD(DD / 256, BT / 128);              // (4, 32)
    const dim3 gF(FF / 256, BT / 128);              // (16, 32)
    const dim3 gV(VV / 256, BT / 128);              // (125, 32)

    embed_kernel<<<(BT * DD + 255) / 256, blk, 0, stream>>>(idx, tok, pos, x);

    for (int l = 0; l < LL; ++l) {
        const size_t oDD = (size_t)l * DD * DD;
        const size_t oDF = (size_t)l * DD * FF;
        const size_t oD  = (size_t)l * DD;
        const size_t oF  = (size_t)l * FF;

        // convert+transpose this layer's weights to bf16 (N x K)
        convT_kernel<<<dim3(DD / 32, DD / 32), blk, 0, stream>>>(Wq + oDD, wq_t, DD, DD);
        convT_kernel<<<dim3(DD / 32, DD / 32), blk, 0, stream>>>(Wk + oDD, wk_t, DD, DD);
        convT_kernel<<<dim3(DD / 32, DD / 32), blk, 0, stream>>>(Wv + oDD, wv_t, DD, DD);
        convT_kernel<<<dim3(DD / 32, DD / 32), blk, 0, stream>>>(Wo + oDD, wo_t, DD, DD);
        convT_kernel<<<dim3(FF / 32, DD / 32), blk, 0, stream>>>(W1 + oDF, w1_t, DD, FF);
        convT_kernel<<<dim3(DD / 32, FF / 32), blk, 0, stream>>>(W2 + oDF, w2_t, FF, DD);

        ln_kernel<<<BT, blk, 0, stream>>>(x, ln1g + oD, ln1b + oD, hb);
        gemm_bf16_wmma<true, false, false><<<gD, blk, 0, stream>>>(
            hb, wq_t, bq + oD, nullptr, (void*)qf, BT, DD, DD);
        gemm_bf16_wmma<true, false, false><<<gD, blk, 0, stream>>>(
            hb, wk_t, bk + oD, nullptr, (void*)kf, BT, DD, DD);
        gemm_bf16_wmma<true, false, false><<<gD, blk, 0, stream>>>(
            hb, wv_t, bv + oD, nullptr, (void*)vf, BT, DD, DD);
        band_attn_kernel<<<(BT * HH + 255) / 256, blk, 0, stream>>>(qf, kf, vf, yb);
        // x = x + y @ Wo + bo
        gemm_bf16_wmma<true, true, false><<<gD, blk, 0, stream>>>(
            yb, wo_t, bo + oD, x, (void*)x, BT, DD, DD);

        ln_kernel<<<BT, blk, 0, stream>>>(x, ln2g + oD, ln2b + oD, hb);
        // m1 = gelu(h @ W1 + b1), bf16 out (fused epilogue)
        gemm_bf16_wmma<true, false, true><<<gF, blk, 0, stream>>>(
            hb, w1_t, b1 + oF, nullptr, (void*)m1b, BT, FF, DD);
        // x = x + m1 @ W2 + b2
        gemm_bf16_wmma<true, true, false><<<gD, blk, 0, stream>>>(
            m1b, w2_t, b2 + oD, x, (void*)x, BT, DD, FF);
    }

    ln_kernel<<<BT, blk, 0, stream>>>(x, lnfg, lnfb, hb);
    convT_kernel<<<dim3(VV / 32, DD / 32), blk, 0, stream>>>(lmW, wt, DD, VV);
    gemm_bf16_wmma<false, false, false><<<gV, blk, 0, stream>>>(
        hb, wt, nullptr, nullptr, (void*)out, BT, VV, DD);
}
